// UniGCNII_pyg_64811056496748
// MI455X (gfx1250) — matrix operand order repaired and checked by reading, not compiled, not run on hardware
//
#include <hip/hip_runtime.h>
#include <cmath>

#define N_V   100000
#define M_E   50000
#define NNZ_  800000
#define F     128
#define NCLS  40
#define DEPTH 4
#define ALPHA 0.1f

typedef __attribute__((ext_vector_type(8)))  _Float16 v8h;
typedef __attribute__((ext_vector_type(16))) _Float16 v16h;
typedef __attribute__((ext_vector_type(8)))  float    v8f;

// ---------------------------------------------------------------- utilities
__global__ __launch_bounds__(256) void k_zero(float* __restrict__ p, long n4) {
  long i = (long)blockIdx.x * 256 + threadIdx.x;
  if (i < n4) ((float4*)p)[i] = make_float4(0.f, 0.f, 0.f, 0.f);
}

__global__ __launch_bounds__(256) void k_count(const int* __restrict__ vtx,
                                               const int* __restrict__ edg,
                                               float* __restrict__ cv,
                                               float* __restrict__ ce) {
  int t = blockIdx.x * 256 + threadIdx.x;
  if (t < NNZ_) {
    atomicAdd(cv + vtx[t], 1.0f);
    atomicAdd(ce + edg[t], 1.0f);
  }
}

__global__ __launch_bounds__(256) void k_rcp(float* __restrict__ p, int n) {
  int t = blockIdx.x * 256 + threadIdx.x;
  if (t < n) p[t] = 1.0f / fmaxf(p[t], 1.0f);
}

// One wave per incidence entry: gather 128 floats (float4/lane, coalesced),
// optional per-row scale, scatter with global_atomic_add_f32 (L2-resident dst).
__global__ __launch_bounds__(256) void k_scatter(const float* __restrict__ src,
                                                 const int* __restrict__ gIdx,
                                                 const int* __restrict__ sIdx,
                                                 const float* __restrict__ scale,
                                                 float* __restrict__ dst) {
  int wid  = blockIdx.x * 8 + (threadIdx.x >> 5);
  int lane = threadIdx.x & 31;
  if (wid >= NNZ_) return;
  int g = gIdx[wid];
  int s = sIdx[wid];
  float sc = scale ? scale[g] : 1.0f;
  float4 v = *(const float4*)(src + (size_t)g * F + lane * 4);
  float* d = dst + (size_t)s * F + lane * 4;
  atomicAdd(d + 0, v.x * sc);
  atomicAdd(d + 1, v.y * sc);
  atomicAdd(d + 2, v.z * sc);
  atomicAdd(d + 3, v.w * sc);
}

// ---------------------------------------------------------------- fused GEMM
// MODE 0: out = relu(A @ W + bias)                       (input projection)
// MODE 1: Xi = 0.9*normalize(A*rcp_v) + 0.1*x0 ; out = relu((1-b)Xi + b*(Xi@W))
// MODE 2: out = A @ W + bias   (W is 128x40, padded to 48 cols)
template <int MODE>
__global__ __launch_bounds__(256) void k_gemm(const float* __restrict__ A,
                                              const float* __restrict__ W,
                                              const float* __restrict__ bias,
                                              const float* __restrict__ x0,
                                              const float* __restrict__ rv,
                                              float beta,
                                              float* __restrict__ out) {
  constexpr int NC  = (MODE == 2) ? 48 : 128;  // padded output cols
  constexpr int NO  = (MODE == 2) ? 40 : 128;  // real output cols
  constexpr int LDW = 136;                     // f16 row pad: 272B -> bank stride 4
  constexpr int LDA = 132;                     // f32 row pad

  __shared__ __attribute__((aligned(16))) _Float16 sWt[NC * LDW];  // W^T, f16
  __shared__ __attribute__((aligned(16))) _Float16 sXh[16 * LDW];  // A tile, f16
  __shared__ float sXi[16 * LDA];                                  // A tile, f32
  __shared__ float sRed[256];
  __shared__ float sScale[16];

  const int tid = threadIdx.x;
  const int r0  = blockIdx.x * 16;

  // ---- stage W transposed (f32 -> f16), coalesced global reads
  if (MODE == 2) {
    for (int i = tid; i < 128 * 48; i += 256) {
      int k = i / 48, n = i % 48;
      float w = (n < NCLS) ? W[k * NCLS + n] : 0.0f;
      sWt[n * LDW + k] = (_Float16)w;
    }
  } else {
    for (int i = tid; i < 128 * 128; i += 256) {
      int k = i >> 7, n = i & 127;
      sWt[n * LDW + k] = (_Float16)W[(k << 7) + n];
    }
  }

  // ---- stage A tile (16 rows x 128); 16 threads/row, 8 cols each
  const int row = tid >> 4;
  const int seg = (tid & 15) << 3;
  if (MODE == 1) {
    float rvv = rv[r0 + row];
    const float4* a4 = (const float4*)(A + (size_t)(r0 + row) * F + seg);
    float4 p0 = a4[0], p1 = a4[1];
    float vals[8] = {p0.x * rvv, p0.y * rvv, p0.z * rvv, p0.w * rvv,
                     p1.x * rvv, p1.y * rvv, p1.z * rvv, p1.w * rvv};
    float ss = 0.f;
#pragma unroll
    for (int j = 0; j < 8; ++j) ss += vals[j] * vals[j];
    sRed[tid] = ss;
    __syncthreads();
    if (tid < 16) {  // row L2-norm + (1-alpha)/norm scale
      float s = 0.f;
#pragma unroll
      for (int j = 0; j < 16; ++j) s += sRed[tid * 16 + j];
      float nrm = sqrtf(s);
      sScale[tid] = (nrm > 0.0f) ? (1.0f - ALPHA) / nrm : 0.0f;
    }
    __syncthreads();
    float sc = sScale[row];
    const float4* x4 = (const float4*)(x0 + (size_t)(r0 + row) * F + seg);
    float4 q0 = x4[0], q1 = x4[1];
    float xs[8] = {q0.x, q0.y, q0.z, q0.w, q1.x, q1.y, q1.z, q1.w};
#pragma unroll
    for (int j = 0; j < 8; ++j) {
      float xi = vals[j] * sc + ALPHA * xs[j];   // Xi = 0.9*Xv/|Xv| + 0.1*x0
      sXi[row * LDA + seg + j] = xi;
      sXh[row * LDW + seg + j] = (_Float16)xi;
    }
  } else {
    const float4* a4 = (const float4*)(A + (size_t)(r0 + row) * F + seg);
    float4 p0 = a4[0], p1 = a4[1];
    float xs[8] = {p0.x, p0.y, p0.z, p0.w, p1.x, p1.y, p1.z, p1.w};
#pragma unroll
    for (int j = 0; j < 8; ++j) sXh[row * LDW + seg + j] = (_Float16)xs[j];
  }
  __syncthreads();

  // ---- WMMA: each wave owns one 16-col tile; K loop = 4 x (16x16x32)
  const int wv   = tid >> 5;
  const int lane = tid & 31;
  const int hf   = lane >> 4;   // lane half
  const int ln   = lane & 15;   // M (for A) / N (for B,C,D)
  const int n0   = wv * 16;
  if (n0 < NC) {
    v8f acc = {};
    const _Float16* ap = &sXh[ln * LDW];
    const _Float16* bp = &sWt[(n0 + ln) * LDW];
#pragma unroll
    for (int kk = 0; kk < 4; ++kk) {
      // A frag (16-bit A layout): K = kk*32 + {j+8h (j<8), 8+j+8h (j>=8)}
      v8h a0 = *(const v8h*)(ap + kk * 32 + 8 * hf);
      v8h a1 = *(const v8h*)(ap + kk * 32 + 8 * hf + 16);
      // B frag (16-bit B layout): K = kk*32 + {j+8h (j<8), 8+j+8h (j>=8)}
      v8h b0 = *(const v8h*)(bp + kk * 32 + 8 * hf);
      v8h b1 = *(const v8h*)(bp + kk * 32 + 8 * hf + 16);
      v16h av, bv;
#pragma unroll
      for (int j = 0; j < 8; ++j) {
        av[j] = a0[j]; av[j + 8] = a1[j];
        bv[j] = b0[j]; bv[j + 8] = b1[j];
      }
      acc = __builtin_amdgcn_wmma_f32_16x16x32_f16(false, av, false, bv,
                                                   (short)0, acc, false, false);
    }
    // ---- epilogue; C/D layout: VGPR r holds M = r + 8*half, N = lane%16
#pragma unroll
    for (int r = 0; r < 8; ++r) {
      int orow = r + 8 * hf;
      int col  = n0 + ln;
      if (MODE == 2 && col >= NO) continue;
      float v = acc[r];
      float o;
      if (MODE == 0)      o = fmaxf(v + bias[col], 0.0f);
      else if (MODE == 1) o = fmaxf((1.0f - beta) * sXi[orow * LDA + col] + beta * v, 0.0f);
      else                o = v + bias[col];
      out[(size_t)(r0 + orow) * NO + col] = o;
    }
  }
}

// ---------------------------------------------------------------- launcher
extern "C" void kernel_launch(void* const* d_in, const int* in_sizes, int n_in,
                              void* d_out, int out_size, void* d_ws, size_t ws_size,
                              hipStream_t stream) {
  (void)in_sizes; (void)n_in; (void)out_size; (void)ws_size;
  const float* x    = (const float*)d_in[0];
  const float* W0   = (const float*)d_in[1];
  const float* b0   = (const float*)d_in[2];
  const float* Ws   = (const float*)d_in[3];
  const float* Wout = (const float*)d_in[4];
  const float* bout = (const float*)d_in[5];
  const int*   vtx  = (const int*)d_in[6];
  const int*   edg  = (const int*)d_in[7];
  float* out = (float*)d_out;

  float* ws   = (float*)d_ws;
  float* x0   = ws;                                          // [N,128]
  float* xb   = ws + (size_t)N_V * F;                        // [N,128]
  float* Xv   = ws + (size_t)2 * N_V * F;                    // [N,128]
  float* Xe   = ws + (size_t)3 * N_V * F;                    // [M,128]
  float* rcpv = ws + (size_t)3 * N_V * F + (size_t)M_E * F;  // [N]
  float* rcpe = rcpv + N_V;                                  // [M]

  auto z4 = [&](float* p, long n) {
    long n4 = n / 4;
    k_zero<<<(n4 + 255) / 256, 256, 0, stream>>>(p, n4);
  };

  // degree counts -> reciprocal of max(count,1)
  z4(rcpv, N_V);
  z4(rcpe, M_E);
  k_count<<<(NNZ_ + 255) / 256, 256, 0, stream>>>(vtx, edg, rcpv, rcpe);
  k_rcp<<<(N_V + 255) / 256, 256, 0, stream>>>(rcpv, N_V);
  k_rcp<<<(M_E + 255) / 256, 256, 0, stream>>>(rcpe, M_E);

  // x0 = relu(x @ W0 + b0)
  k_gemm<0><<<N_V / 16, 256, 0, stream>>>(x, W0, b0, nullptr, nullptr, 0.f, x0);

  for (int i = 0; i < DEPTH; ++i) {
    const float* cur = (i == 0) ? x0 : xb;
    z4(Xe, (long)M_E * F);
    k_scatter<<<NNZ_ / 8, 256, 0, stream>>>(cur, vtx, edg, nullptr, Xe);   // v->e sum
    z4(Xv, (long)N_V * F);
    k_scatter<<<NNZ_ / 8, 256, 0, stream>>>(Xe, edg, vtx, rcpe, Xv);       // e->v (Xe/cnt_e)
    float beta = logf(0.5f / (float)(i + 1) + 1.0f);
    k_gemm<1><<<N_V / 16, 256, 0, stream>>>(Xv, Ws + (size_t)i * F * F,
                                            nullptr, x0, rcpv, beta, xb);
  }
  // out = x @ Wout + bout
  k_gemm<2><<<N_V / 16, 256, 0, stream>>>(xb, Wout, bout, nullptr, nullptr, 0.f, out);
}
